// plain_GNN_85512798863577
// MI455X (gfx1250) — matrix operand order repaired and checked by reading, not compiled, not run on hardware
//
#include <hip/hip_runtime.h>
#include <hip/hip_bf16.h>

// ---------------------------------------------------------------------------
// GCN forward on gfx1250 (MI455X): bf16 WMMA GEMM + atomic SpMM aggregation.
// ---------------------------------------------------------------------------

typedef __bf16 v16bf __attribute__((ext_vector_type(16)));
typedef __bf16 v8bf  __attribute__((ext_vector_type(8)));
typedef __bf16 v4bf  __attribute__((ext_vector_type(4)));
typedef float  v8f   __attribute__((ext_vector_type(8)));

#define NFEAT 256

// ----------------------------- normalization -------------------------------

__global__ __launch_bounds__(256) void gcn_init_deg(float* __restrict__ deg, int n) {
    int i = blockIdx.x * 256 + threadIdx.x;
    if (i < n) deg[i] = 1.0f;                     // self-loop contributes 1
}

__global__ __launch_bounds__(256) void gcn_accum_deg(const int* __restrict__ dst,
                                                     float* __restrict__ deg, int e) {
    int i = blockIdx.x * 256 + threadIdx.x;
    if (i < e) atomicAdd(&deg[dst[i]], 1.0f);
}

__global__ __launch_bounds__(256) void gcn_dinv(const float* __restrict__ deg,
                                                float* __restrict__ dinv, int n) {
    int i = blockIdx.x * 256 + threadIdx.x;
    if (i < n) dinv[i] = rsqrtf(deg[i]);          // deg >= 1 always
}

// ------------------------------- WMMA GEMM ---------------------------------
// H[N x M] = A[N x K] * W[K x M], f32 in/out, bf16 multiply, f32 accumulate.
// Block: 128 threads (4 waves), 64x64 output tile, wave -> 32x32 (2x2 WMMA).
// Separate LDS arrays (no union aliasing) so the backend keeps ds_load_b128
// for the fragment loads; C staged through its own LDS tile for coalesced
// float4 global stores. 24KB LDS / block -> no occupancy impact on a WGP.
__global__ __launch_bounds__(128) void gcn_gemm_bf16(
    const float* __restrict__ A, const float* __restrict__ W,
    float* __restrict__ H, int Nrows, int K, int M)
{
    __shared__ __bf16 lds_a[64][32];   // [row][k]
    __shared__ __bf16 lds_b[64][32];   // [col][k] (transposed during load)
    __shared__ float  lds_c[64][64];   // output tile

    const int tid     = threadIdx.x;
    const int lane    = tid & 31;
    const int wave    = tid >> 5;
    const int waveRow = wave >> 1;     // 0..1
    const int waveCol = wave & 1;      // 0..1
    const int rowBase = blockIdx.y * 64;
    const int colBase = blockIdx.x * 64;
    const int rMax    = Nrows - 1;

    // B staging role: one column per thread, 16 consecutive K values.
    const int bCol = tid & 63;
    const int bKb  = (tid >> 6) * 16;   // 0 | 16

    v8f acc[2][2];
#pragma unroll
    for (int m = 0; m < 2; ++m)
#pragma unroll
        for (int n = 0; n < 2; ++n)
#pragma unroll
            for (int i = 0; i < 8; ++i)
                acc[m][n][i] = 0.0f;

    for (int k0 = 0; k0 < K; k0 += 32) {
        // ---- A tile 64x32: float4 loads (clamped row -> no predication),
        //      pack 4xbf16 -> ds_store_b64 (merged to b128 by compiler).
#pragma unroll
        for (int i = 0; i < 4; ++i) {
            int g = tid + i * 128;              // 0..511 float4-groups
            int r = g >> 3;                     // 0..63
            int k = (g & 7) * 4;                // 0..28
            int gr = rowBase + r;
            gr = (gr < rMax) ? gr : rMax;       // clamp: always-valid address
            const float4 v = *(const float4*)(A + (size_t)gr * K + k0 + k);
            v4bf p;
            p[0] = (__bf16)v.x; p[1] = (__bf16)v.y;
            p[2] = (__bf16)v.z; p[3] = (__bf16)v.w;
            *(v4bf*)&lds_a[r][k] = p;
        }
        // ---- B tile 32x64: lanes hold consecutive columns (coalesced reads),
        //      each thread owns lds_b[col][kb..kb+15].
        {
            float tmp[16];
#pragma unroll
            for (int kk = 0; kk < 16; ++kk)
                tmp[kk] = W[(size_t)(k0 + bKb + kk) * M + colBase + bCol];
#pragma unroll
            for (int j = 0; j < 4; ++j) {
                v4bf p;
                p[0] = (__bf16)tmp[4 * j + 0]; p[1] = (__bf16)tmp[4 * j + 1];
                p[2] = (__bf16)tmp[4 * j + 2]; p[3] = (__bf16)tmp[4 * j + 3];
                *(v4bf*)&lds_b[bCol][bKb + 4 * j] = p;
            }
        }
        __syncthreads();

        // Fragments in contiguous k order (k = 16*half + s, same for A and B), loaded as explicit vectors
        // (2x ds_load_b128 each).
        const int half = (lane >> 4);           // 0 | 1
        const int kbA  = half * 16;             // A: K 0-15 vs 16-31
        const int kbB  = half * 16;             // B: K 0-15 vs 16-31
        const int l15  = lane & 15;

        v16bf afrag[2], bfrag[2];
#pragma unroll
        for (int m = 0; m < 2; ++m) {
            int row = waveRow * 32 + m * 16 + l15;
            const v8bf lo = *(const v8bf*)&lds_a[row][kbA];       // 16B aligned
            const v8bf hi = *(const v8bf*)&lds_a[row][kbA + 8];   // 16B aligned
            afrag[m] = __builtin_shufflevector(lo, hi,
                0, 1, 2, 3, 4, 5, 6, 7, 8, 9, 10, 11, 12, 13, 14, 15);
        }
#pragma unroll
        for (int n = 0; n < 2; ++n) {
            int col = waveCol * 32 + n * 16 + l15;
            bfrag[n] = *(const v16bf*)&lds_b[col][kbB];           // 32B aligned
        }

#pragma unroll
        for (int m = 0; m < 2; ++m)
#pragma unroll
            for (int n = 0; n < 2; ++n)
                acc[m][n] = __builtin_amdgcn_wmma_f32_16x16x32_bf16(
                    /*neg_a=*/false, afrag[m],
                    /*neg_b=*/false, bfrag[n],
                    /*c_mod=*/(short)0, acc[m][n],
                    /*reuse_a=*/false, /*reuse_b=*/false);
        __syncthreads();
    }

    // ---- Epilogue: deposit C into LDS tile, then coalesced float4 stores.
    // C/D layout: VGPR v -> row ((lane>=16?8:0)+v), col (lane&15) of 16x16 tile.
    const int rsub = (lane >> 4) * 8;
    const int ncol = lane & 15;
#pragma unroll
    for (int m = 0; m < 2; ++m)
#pragma unroll
        for (int n = 0; n < 2; ++n)
#pragma unroll
            for (int v = 0; v < 8; ++v)
                lds_c[waveRow * 32 + m * 16 + rsub + v]
                     [waveCol * 32 + n * 16 + ncol] = acc[m][n][v];
    __syncthreads();

    const bool full = (rowBase + 64 <= Nrows);   // block-uniform fast path
#pragma unroll
    for (int i = 0; i < 8; ++i) {
        int g = tid + i * 128;                   // 0..1023 float4-groups
        int r = g >> 4;                          // 0..63
        int c = (g & 15) * 4;                    // 0..60
        if (full || (rowBase + r) < Nrows) {
            float4 v = *(const float4*)&lds_c[r][c];
            *(float4*)(H + (size_t)(rowBase + r) * M + colBase + c) = v;
        }
    }
}

// ------------------------------ aggregation --------------------------------

__global__ __launch_bounds__(256) void gcn_zero4(float4* __restrict__ p, long long n4) {
    long long i = (long long)blockIdx.x * 256 + threadIdx.x;
    if (i < n4) p[i] = make_float4(0.f, 0.f, 0.f, 0.f);
}

// One edge per wave32; lanes stride features -> 128B coalesced gather,
// per-float global_atomic_add_f32 scatter (bandwidth-bound, L2 atomics).
template <int M>
__global__ __launch_bounds__(256) void gcn_scatter(
    const float* __restrict__ H, const int* __restrict__ src,
    const int* __restrict__ dst, const float* __restrict__ dinv,
    float* __restrict__ ACC, int E)
{
    const int lane = threadIdx.x & 31;
    const int e = blockIdx.x * 8 + (threadIdx.x >> 5);
    if (e >= E) return;
    const int s = src[e];
    const int d = dst[e];
    const float w = dinv[s] * dinv[d];
    const float* __restrict__ hs = H + (size_t)s * M;
    float* __restrict__ ad = ACC + (size_t)d * M;
#pragma unroll
    for (int j = 0; j < M / 32; ++j) {
        int f = lane + 32 * j;
        atomicAdd(&ad[f], w * hs[f]);
    }
}

// out = relu(ACC + dinv[i]^2 * H + bias), float4-vectorized.
template <int M>
__global__ __launch_bounds__(256) void gcn_finalize(
    const float* __restrict__ ACC, const float* __restrict__ H,
    const float* __restrict__ dinv, const float* __restrict__ bias,
    float* __restrict__ out, long long total4)
{
    long long g = (long long)blockIdx.x * 256 + threadIdx.x;
    if (g >= total4) return;
    long long base = g * 4;
    int i = (int)(base / M);        // M is a power of two -> shift
    int f = (int)(base % M);
    float4 a = ((const float4*)ACC)[g];
    float4 h = ((const float4*)H)[g];
    float4 b = *(const float4*)(bias + f);
    float s = dinv[i]; s *= s;
    float4 r;
    r.x = fmaxf(a.x + s * h.x + b.x, 0.f);
    r.y = fmaxf(a.y + s * h.y + b.y, 0.f);
    r.z = fmaxf(a.z + s * h.z + b.z, 0.f);
    r.w = fmaxf(a.w + s * h.w + b.w, 0.f);
    ((float4*)out)[g] = r;
}

// ------------------------------- launcher ----------------------------------

extern "C" void kernel_launch(void* const* d_in, const int* in_sizes, int n_in,
                              void* d_out, int out_size, void* d_ws, size_t ws_size,
                              hipStream_t stream)
{
    const float* x      = (const float*)d_in[0];
    const float* W_enc  = (const float*)d_in[1];
    const float* b_enc  = (const float*)d_in[2];
    const float* W_conv = (const float*)d_in[3];
    const float* b_conv = (const float*)d_in[4];
    const float* W_dec  = (const float*)d_in[5];
    const float* b_dec  = (const float*)d_in[6];
    const int*   edge   = (const int*)d_in[7];

    const int N = in_sizes[0] / NFEAT;      // 100000
    const int E = in_sizes[7] / 2;          // 1600000
    const int* src = edge;
    const int* dst = edge + E;

    // Workspace carve-up (floats): X | H | ACC | deg | dinv
    float* Xbuf = (float*)d_ws;
    float* Hbuf = Xbuf + (size_t)N * NFEAT;
    float* ACC  = Hbuf + (size_t)N * NFEAT;
    float* deg  = ACC  + (size_t)N * NFEAT;
    float* dinv = deg + N;

    // Symmetric GCN normalization (recomputed each call: deterministic).
    gcn_init_deg <<<(N + 255) / 256, 256, 0, stream>>>(deg, N);
    gcn_accum_deg<<<(E + 255) / 256, 256, 0, stream>>>(dst, deg, E);
    gcn_dinv     <<<(N + 255) / 256, 256, 0, stream>>>(deg, dinv, N);

    struct Layer { const float* W; const float* b; int M; };
    const Layer layers[6] = {
        {W_enc,  b_enc,  256},
        {W_conv, b_conv, 256}, {W_conv, b_conv, 256},
        {W_conv, b_conv, 256}, {W_conv, b_conv, 256},
        {W_dec,  b_dec,  64},
    };

    const float* Ain = x;
    int K = NFEAT;
    for (int l = 0; l < 6; ++l) {
        const int M = layers[l].M;
        dim3 grid(M / 64, (N + 63) / 64);
        gcn_gemm_bf16<<<grid, 128, 0, stream>>>(Ain, layers[l].W, Hbuf, N, K, M);

        const long long tot4 = ((long long)N * M) / 4;
        gcn_zero4<<<(unsigned)((tot4 + 255) / 256), 256, 0, stream>>>((float4*)ACC, tot4);

        if (M == 256)
            gcn_scatter<256><<<(E + 7) / 8, 256, 0, stream>>>(Hbuf, src, dst, dinv, ACC, E);
        else
            gcn_scatter<64> <<<(E + 7) / 8, 256, 0, stream>>>(Hbuf, src, dst, dinv, ACC, E);

        float* outp = (l == 5) ? (float*)d_out : Xbuf;
        if (M == 256)
            gcn_finalize<256><<<(unsigned)((tot4 + 255) / 256), 256, 0, stream>>>(
                ACC, Hbuf, dinv, layers[l].b, outp, tot4);
        else
            gcn_finalize<64> <<<(unsigned)((tot4 + 255) / 256), 256, 0, stream>>>(
                ACC, Hbuf, dinv, layers[l].b, outp, tot4);

        Ain = outp;
        K = M;
    }
}
